// SP_SA_6184752906711
// MI455X (gfx1250) — matrix operand order repaired and checked by reading, not compiled, not run on hardware
//
#include <hip/hip_runtime.h>
#include <hip/hip_bf16.h>

typedef __bf16 v16bf __attribute__((ext_vector_type(16)));
typedef __bf16 v8bf  __attribute__((ext_vector_type(8)));
typedef __bf16 v4bf  __attribute__((ext_vector_type(4)));
typedef float  v8f   __attribute__((ext_vector_type(8)));

#define DIMC   768
#define HEADS  12
#define HD     64
#define BATCH  64
#define NTOK   196
#define NROWS  (BATCH * NTOK)      // 12544
#define QSCALE 0.125f              // 64^-0.5

// ---------------------------------------------------------------------------
// WMMA fragment loaders (wave32; A and B both use contiguous k = 16h + s), bf16 sources.
// A (16x32): lane L -> row m=L&15; half h=L>>4; elems 0..7 = K(k0+16h..+7),
//            elems 8..15 = K(k0+16h+8..+15)  -> two contiguous 16B loads.
// B (32x16): lane L -> col n=L&15; half kh;  elems 0..15 = K(k0+16kh..+15)
//            -> one contiguous 32B load.
// C (16x16 f32): lane -> col n=L&15; VGPR r -> row m = r + 8*(L>>4).
// ---------------------------------------------------------------------------
__device__ __forceinline__ v16bf frag_a_lds(const __bf16* s, int lane, int k0) {
  int m = lane & 15, h = lane >> 4;
  const __bf16* p = s + m * DIMC + k0 + 16 * h;
  v8bf lo = *reinterpret_cast<const v8bf*>(p);
  v8bf hi = *reinterpret_cast<const v8bf*>(p + 8);
  v16bf a;
#pragma unroll
  for (int i = 0; i < 8; ++i) { a[i] = lo[i]; a[8 + i] = hi[i]; }
  return a;
}

__device__ __forceinline__ v16bf frag_b_bf16(const __bf16* __restrict__ w, int ld,
                                             int lane, int k0) {
  int n = lane & 15, kh = lane >> 4;
  return *reinterpret_cast<const v16bf*>(w + (size_t)n * ld + k0 + 16 * kh);
}

// Async DMA of one contiguous 16-row x 768-col bf16 slab (24576 B) into LDS.
// 1536 chunks of 16 B; 256 threads issue 6 async copies each.
__device__ __forceinline__ void stage_a_async(const __bf16* g, __bf16* s) {
  uint32_t lds_base = (uint32_t)(uintptr_t)s;        // low 32 bits = LDS offset
  uint64_t gbase    = (uint64_t)(uintptr_t)g;
#pragma unroll
  for (int c = 0; c < 6; ++c) {
    uint32_t byte_off = (uint32_t)(threadIdx.x + c * 256) * 16u;
    uint32_t ldsa = lds_base + byte_off;
    uint64_t ga   = gbase + byte_off;
    asm volatile("global_load_async_to_lds_b128 %0, %1, off"
                 :: "v"(ldsa), "v"(ga) : "memory");
  }
  asm volatile("s_wait_asynccnt 0" ::: "memory");
  __syncthreads();
}

// ---------------------------------------------------------------------------
// Kernel 0: fp32 -> bf16 conversion (vector of 4 per thread)
// ---------------------------------------------------------------------------
__global__ __launch_bounds__(256) void cvt_kernel(const float* __restrict__ src,
                                                  __bf16* __restrict__ dst, int n4) {
  int i = blockIdx.x * 256 + threadIdx.x;
  if (i < n4) {
    float4 f = reinterpret_cast<const float4*>(src)[i];
    v4bf o;
    o[0] = (__bf16)f.x; o[1] = (__bf16)f.y; o[2] = (__bf16)f.z; o[3] = (__bf16)f.w;
    reinterpret_cast<v4bf*>(dst)[i] = o;
  }
}

// ---------------------------------------------------------------------------
// Kernel 1: relative-position bias map  pos_map[h][i][j]  (H,N,N) fp32
// ---------------------------------------------------------------------------
__global__ __launch_bounds__(256) void pos_kernel(const float* __restrict__ w1_w,
                                                  const float* __restrict__ w1_b,
                                                  const float* __restrict__ w2,
                                                  const float* __restrict__ b2,
                                                  float* __restrict__ pm) {
  __shared__ float s_w1[2 * DIMC];
  __shared__ float s_b1[DIMC];
  __shared__ float s_w2[DIMC];
  for (int i = threadIdx.x; i < 2 * DIMC; i += 256) s_w1[i] = w1_w[i];
  for (int i = threadIdx.x; i < DIMC; i += 256) { s_b1[i] = w1_b[i]; s_w2[i] = w2[i]; }
  __syncthreads();

  int idx = blockIdx.x * 256 + threadIdx.x;
  if (idx >= NTOK * NTOK) return;
  int i = idx / NTOK, j = idx % NTOK;
  float indx = (float)((j % 14) - (i % 14));
  float indy = (float)((j / 14) - (i / 14));

  for (int h = 0; h < HEADS; ++h) {
    float acc = 0.f;
    for (int d = 0; d < HD; ++d) {
      int c = h * HD + d;
      float e = fmaxf(indx * s_w1[2 * c] + indy * s_w1[2 * c + 1] + s_b1[c], 0.f);
      acc += e * s_w2[c];
    }
    pm[(size_t)h * NTOK * NTOK + idx] = acc + b2[h];
  }
}

// ---------------------------------------------------------------------------
// Kernel 2: fused QKV projection GEMM (12544 x 2304 x 768), bf16 WMMA.
// One block per 16-row slab: A staged in LDS via async DMA; 8 waves stride
// over the 18 column tiles of 128.  Q scaled by QSCALE at store.
// q/k/v out: (B, H, N, HD) bf16.
// ---------------------------------------------------------------------------
__global__ __launch_bounds__(256) void qkv_kernel(const __bf16* __restrict__ xb,
                                                  const __bf16* __restrict__ qkwb,
                                                  const __bf16* __restrict__ vwb,
                                                  __bf16* __restrict__ q_ws,
                                                  __bf16* __restrict__ k_ws,
                                                  __bf16* __restrict__ v_ws) {
  const int NT = (2 * DIMC + DIMC) / 128;  // 18 column tiles of 128
  __shared__ __bf16 s_a[16 * DIMC];        // 24 KB
  int lane = threadIdx.x & 31, wave = threadIdx.x >> 5;
  int mt = blockIdx.x;
  int half = lane >> 4, ln = lane & 15;

  stage_a_async(xb + (size_t)mt * 16 * DIMC, s_a);

  for (int nt = wave; nt < NT; nt += 8) {
    int n0 = nt * 128;
    const __bf16* wsrc;
    int wn0;
    if (n0 < 2 * DIMC) { wsrc = qkwb; wn0 = n0; }
    else               { wsrc = vwb;  wn0 = n0 - 2 * DIMC; }

    v8f acc[8];
#pragma unroll
    for (int j = 0; j < 8; ++j)
#pragma unroll
      for (int r = 0; r < 8; ++r) acc[j][r] = 0.f;

    for (int k0 = 0; k0 < DIMC; k0 += 32) {
      if (k0 + 32 < DIMC)
        __builtin_prefetch(wsrc + (size_t)(wn0 + ln) * DIMC + k0 + 32, 0, 1);
      v16bf a = frag_a_lds(s_a, lane, k0);
#pragma unroll
      for (int j = 0; j < 8; ++j) {
        v16bf b = frag_b_bf16(wsrc + (size_t)(wn0 + 16 * j) * DIMC, DIMC, lane, k0);
        acc[j] = __builtin_amdgcn_wmma_f32_16x16x32_bf16(false, a, false, b,
                                                         (short)0, acc[j], false, false);
      }
    }

#pragma unroll
    for (int j = 0; j < 8; ++j) {
      int col = n0 + 16 * j + ln;
#pragma unroll
      for (int r = 0; r < 8; ++r) {
        int row = mt * 16 + r + 8 * half;
        int b_  = row / NTOK, tok = row % NTOK;
        float val = acc[j][r];
        if (col < 2 * DIMC) {
          int pair = col / DIMC;           // 0=q, 1=k
          int cc   = col % DIMC;
          int h = cc / HD, d = cc % HD;
          if (!pair) val *= QSCALE;
          __bf16* dst = pair ? k_ws : q_ws;
          dst[(((size_t)b_ * HEADS + h) * NTOK + tok) * HD + d] = (__bf16)val;
        } else {
          int cc = col - 2 * DIMC;
          int h = cc / HD, d = cc % HD;
          v_ws[(((size_t)b_ * HEADS + h) * NTOK + tok) * HD + d] = (__bf16)val;
        }
      }
    }
  }
}

// ---------------------------------------------------------------------------
// Kernel 3: flash attention per (b, h, 16-query tile); 32-key chunks.
// score = (q*SCALE)@k^T * pos_map -> online softmax -> P@V.
// P transposed from C-layout to A-layout through LDS (ds ops + s_wait_dscnt).
// Output o_ws: (B, N, H*HD) bf16.
// ---------------------------------------------------------------------------
__global__ __launch_bounds__(32) void attn_kernel(const __bf16* __restrict__ q_ws,
                                                  const __bf16* __restrict__ k_ws,
                                                  const __bf16* __restrict__ v_ws,
                                                  const float* __restrict__ pm_ws,
                                                  __bf16* __restrict__ o_ws) {
  const int QT = (NTOK + 15) / 16;  // 13 query tiles
  int bid = blockIdx.x;
  int qt = bid % QT;
  int bh = bid / QT;
  int h = bh % HEADS, b = bh / HEADS;
  int lane = threadIdx.x;
  int half = lane >> 4, ln = lane & 15;

  const __bf16* qb = q_ws + (size_t)bh * NTOK * HD;
  const __bf16* kb = k_ws + (size_t)bh * NTOK * HD;
  const __bf16* vb = v_ws + (size_t)bh * NTOK * HD;
  const float*  pm = pm_ws + (size_t)h * NTOK * NTOK;

  __shared__ __bf16 lds_p[16 * 32];

  // Q fragments (A layout) for K-dim chunks 0..31 and 32..63; clamp padded rows.
  int qrow_base = qt * 16;
  int arow = qrow_base + ln; if (arow > NTOK - 1) arow = NTOK - 1;
  v16bf qa[2];
#pragma unroll
  for (int c = 0; c < 2; ++c) {
    const __bf16* p = qb + (size_t)arow * HD + c * 32 + 16 * half;
    v8bf lo = *reinterpret_cast<const v8bf*>(p);
    v8bf hi = *reinterpret_cast<const v8bf*>(p + 8);
#pragma unroll
    for (int i = 0; i < 8; ++i) { qa[c][i] = lo[i]; qa[c][8 + i] = hi[i]; }
  }

  float m_i[8], l_i[8];
  v8f acc[4];
#pragma unroll
  for (int r = 0; r < 8; ++r) { m_i[r] = -1e30f; l_i[r] = 0.f; }
#pragma unroll
  for (int dt = 0; dt < 4; ++dt)
#pragma unroll
    for (int r = 0; r < 8; ++r) acc[dt][r] = 0.f;

  for (int j0 = 0; j0 < NTOK; j0 += 32) {
    float s[2][8];
#pragma unroll
    for (int jt = 0; jt < 2; ++jt) {
      int key = j0 + 16 * jt + ln;
      int keyc = key < NTOK ? key : NTOK - 1;
      v8f st;
#pragma unroll
      for (int r = 0; r < 8; ++r) st[r] = 0.f;
#pragma unroll
      for (int c = 0; c < 2; ++c) {
        // B layout: lane = key column, K = head-dim -> 16 contiguous bf16
        v16bf kf = *reinterpret_cast<const v16bf*>(
            kb + (size_t)keyc * HD + c * 32 + 16 * half);
        st = __builtin_amdgcn_wmma_f32_16x16x32_bf16(false, qa[c], false, kf,
                                                     (short)0, st, false, false);
      }
#pragma unroll
      for (int r = 0; r < 8; ++r) {
        int qrow = qrow_base + r + 8 * half; if (qrow > NTOK - 1) qrow = NTOK - 1;
        float sv = st[r];
        s[jt][r] = (key < NTOK) ? sv * pm[(size_t)qrow * NTOK + key] : -1e30f;
      }
    }

    // online softmax across 32-key chunk (row = 16 lanes in a half x 2 subtiles)
    float alpha[8];
#pragma unroll
    for (int r = 0; r < 8; ++r) {
      float t = fmaxf(s[0][r], s[1][r]);
#pragma unroll
      for (int mk = 1; mk <= 8; mk <<= 1) t = fmaxf(t, __shfl_xor(t, mk, 32));
      float mnew = fmaxf(m_i[r], t);
      alpha[r] = __expf(m_i[r] - mnew);
      m_i[r] = mnew;
      float p0 = __expf(s[0][r] - mnew);
      float p1 = __expf(s[1][r] - mnew);
      s[0][r] = p0; s[1][r] = p1;
      float rs = p0 + p1;
#pragma unroll
      for (int mk = 1; mk <= 8; mk <<= 1) rs += __shfl_xor(rs, mk, 32);
      l_i[r] = l_i[r] * alpha[r] + rs;
    }
#pragma unroll
    for (int dt = 0; dt < 4; ++dt)
#pragma unroll
      for (int r = 0; r < 8; ++r) acc[dt][r] *= alpha[r];

    // P: C-layout -> LDS -> A-layout (16x32 bf16)
    asm volatile("s_wait_dscnt 0" ::: "memory");  // prior-iter reads done before overwrite
#pragma unroll
    for (int jt = 0; jt < 2; ++jt)
#pragma unroll
      for (int r = 0; r < 8; ++r)
        lds_p[(r + 8 * half) * 32 + 16 * jt + ln] = (__bf16)s[jt][r];
    asm volatile("s_wait_dscnt 0" ::: "memory");

    v16bf pa;
    {
      const __bf16* p = &lds_p[ln * 32 + 16 * half];
#pragma unroll
      for (int i = 0; i < 8; ++i) { pa[i] = p[i]; pa[8 + i] = p[8 + i]; }
    }

    // P @ V : 4 d-subtiles of 16 output dims
#pragma unroll
    for (int dt = 0; dt < 4; ++dt) {
      v16bf vf;  // B layout: lane = d column, K = key index within chunk
#pragma unroll
      for (int v = 0; v < 8; ++v) {
        int k = 16 * half + 2 * v;
        int key0 = j0 + k;     if (key0 > NTOK - 1) key0 = NTOK - 1;
        int key1 = j0 + k + 1; if (key1 > NTOK - 1) key1 = NTOK - 1;
        vf[2 * v]     = vb[(size_t)key0 * HD + dt * 16 + ln];
        vf[2 * v + 1] = vb[(size_t)key1 * HD + dt * 16 + ln];
      }
      acc[dt] = __builtin_amdgcn_wmma_f32_16x16x32_bf16(false, pa, false, vf,
                                                        (short)0, acc[dt], false, false);
    }
  }

  // epilogue: out (B, N, H*HD) bf16
#pragma unroll
  for (int r = 0; r < 8; ++r) {
    int row = qrow_base + r + 8 * half;
    if (row < NTOK) {
      float inv = 1.f / l_i[r];
#pragma unroll
      for (int dt = 0; dt < 4; ++dt)
        o_ws[((size_t)b * NTOK + row) * DIMC + h * HD + dt * 16 + ln] =
            (__bf16)(acc[dt][r] * inv);
    }
  }
}

// ---------------------------------------------------------------------------
// Kernel 4: output projection GEMM (12544 x 768 x 768) + bias -> d_out fp32.
// One block per 16-row slab: A staged in LDS via async DMA; 8 waves stride
// over the 12 column tiles of 64.
// ---------------------------------------------------------------------------
__global__ __launch_bounds__(256) void proj_kernel(const __bf16* __restrict__ o_ws,
                                                   const __bf16* __restrict__ pwb,
                                                   const float* __restrict__ proj_b,
                                                   float* __restrict__ out) {
  const int NT = DIMC / 64;  // 12 column tiles of 64
  __shared__ __bf16 s_a[16 * DIMC];  // 24 KB
  int lane = threadIdx.x & 31, wave = threadIdx.x >> 5;
  int mt = blockIdx.x;
  int half = lane >> 4, ln = lane & 15;

  stage_a_async(o_ws + (size_t)mt * 16 * DIMC, s_a);

  for (int nt = wave; nt < NT; nt += 8) {
    int n0 = nt * 64;
    v8f acc[4];
#pragma unroll
    for (int j = 0; j < 4; ++j)
#pragma unroll
      for (int r = 0; r < 8; ++r) acc[j][r] = 0.f;

    for (int k0 = 0; k0 < DIMC; k0 += 32) {
      if (k0 + 32 < DIMC)
        __builtin_prefetch(pwb + (size_t)(n0 + ln) * DIMC + k0 + 32, 0, 1);
      v16bf a = frag_a_lds(s_a, lane, k0);
#pragma unroll
      for (int j = 0; j < 4; ++j) {
        v16bf b = frag_b_bf16(pwb + (size_t)(n0 + 16 * j) * DIMC, DIMC, lane, k0);
        acc[j] = __builtin_amdgcn_wmma_f32_16x16x32_bf16(false, a, false, b,
                                                         (short)0, acc[j], false, false);
      }
    }

#pragma unroll
    for (int j = 0; j < 4; ++j) {
      int col = n0 + 16 * j + ln;
      float bias = proj_b[col];
#pragma unroll
      for (int r = 0; r < 8; ++r) {
        int row = mt * 16 + r + 8 * half;
        out[(size_t)row * DIMC + col] = acc[j][r] + bias;
      }
    }
  }
}

// ---------------------------------------------------------------------------
extern "C" void kernel_launch(void* const* d_in, const int* in_sizes, int n_in,
                              void* d_out, int out_size, void* d_ws, size_t ws_size,
                              hipStream_t stream) {
  const float* x      = (const float*)d_in[0];
  const float* qk_w   = (const float*)d_in[1];
  const float* v_w    = (const float*)d_in[2];
  const float* w1_w   = (const float*)d_in[3];
  const float* w1_b   = (const float*)d_in[4];
  const float* w2     = (const float*)d_in[5];
  const float* b2     = (const float*)d_in[6];
  const float* proj_w = (const float*)d_in[7];
  const float* proj_b = (const float*)d_in[8];
  float* out = (float*)d_out;

  // element counts
  const size_t NX   = (size_t)NROWS * DIMC;         // 9,633,792
  const size_t NQKW = (size_t)2 * DIMC * DIMC;      // 1,179,648
  const size_t NVW  = (size_t)DIMC * DIMC;          //   589,824
  const size_t NPM  = (size_t)HEADS * NTOK * NTOK;  //   460,992

  // workspace layout (all byte offsets 256B-aligned)
  char* ws = (char*)d_ws;
  size_t off = 0;
  __bf16* xb   = (__bf16*)(ws + off); off += NX * 2;        // x bf16
  __bf16* qkwb = (__bf16*)(ws + off); off += NQKW * 2;      // qk_w bf16
  __bf16* vwb  = (__bf16*)(ws + off); off += NVW * 2;       // v_w bf16
  __bf16* pwb  = (__bf16*)(ws + off); off += NVW * 2;       // proj_w bf16
  __bf16* q_ws = (__bf16*)(ws + off); off += NX * 2;
  __bf16* k_ws = (__bf16*)(ws + off); off += NX * 2;
  __bf16* v_ws = (__bf16*)(ws + off); off += NX * 2;
  float*  pm_ws = (float*)(ws + off); off += NPM * 4;
  __bf16* o_ws = (__bf16*)(ws + off); off += NX * 2;

  // 0) fp32 -> bf16 conversions
  cvt_kernel<<<(int)((NX / 4 + 255) / 256), 256, 0, stream>>>(x, xb, (int)(NX / 4));
  cvt_kernel<<<(int)((NQKW / 4 + 255) / 256), 256, 0, stream>>>(qk_w, qkwb, (int)(NQKW / 4));
  cvt_kernel<<<(int)((NVW / 4 + 255) / 256), 256, 0, stream>>>(v_w, vwb, (int)(NVW / 4));
  cvt_kernel<<<(int)((NVW / 4 + 255) / 256), 256, 0, stream>>>(proj_w, pwb, (int)(NVW / 4));

  // 1) positional bias map
  pos_kernel<<<(NTOK * NTOK + 255) / 256, 256, 0, stream>>>(w1_w, w1_b, w2, b2, pm_ws);

  // 2) fused QKV projection: one block per 16-row slab (A in LDS via async DMA)
  qkv_kernel<<<NROWS / 16, 256, 0, stream>>>(xb, qkwb, vwb, q_ws, k_ws, v_ws);

  // 3) flash attention: B*H*13 query tiles, one wave each
  attn_kernel<<<BATCH * HEADS * 13, 32, 0, stream>>>(q_ws, k_ws, v_ws, pm_ws, o_ws);

  // 4) output projection: one block per 16-row slab (A in LDS via async DMA)
  proj_kernel<<<NROWS / 16, 256, 0, stream>>>(o_ws, pwb, proj_b, out);
}
